// HG_GNN_44409961840799
// MI455X (gfx1250) — compile-verified, matrix operand order checked by reading, not run on hardware
//
#include <hip/hip_runtime.h>

#define ITEM_NUM 40000
#define NUM_USERS 10000
#define NNODES (ITEM_NUM + NUM_USERS)   // 50000
#define EM 128
#define NE 800000
#define BS 1024
#define SL 20
#define POSN 200

typedef __attribute__((ext_vector_type(16))) _Float16 v16h;
typedef __attribute__((ext_vector_type(8)))  _Float16 v8h;
typedef __attribute__((ext_vector_type(8)))  float    v8f;

__device__ __forceinline__ float sigmoidf_(float x) {
    return 1.0f / (1.0f + expf(-x));
}

// Load a 16-bit A/B WMMA fragment from K-contiguous f16 memory.
// Per CDNA5 ISA 7.12.2: lane (r,h) element i<8 -> K = h*8+i ; i>=8 -> K = 16+h*8+(i-8).
// base points at (row_or_col, kb); two aligned 16-byte runs.
__device__ __forceinline__ v16h load_frag_f16(const _Float16* __restrict__ base, int h) {
    v8h lo = *(const v8h*)(base + h * 8);
    v8h hi = *(const v8h*)(base + 16 + h * 8);
    v16h r;
    #pragma unroll
    for (int i = 0; i < 8; i++) { r[i] = lo[i]; r[8 + i] = hi[i]; }
    return r;
}

// Same fragment but from f32 memory with a per-row scale (msg/deg path): 2x float4 per run.
__device__ __forceinline__ v16h load_frag_f32(const float* __restrict__ base, int h, float scale) {
    const float4* p = (const float4*)(base + h * 8);
    const float4* q = (const float4*)(base + 16 + h * 8);
    float4 a0 = p[0], a1 = p[1], b0 = q[0], b1 = q[1];
    float lo[8] = {a0.x, a0.y, a0.z, a0.w, a1.x, a1.y, a1.z, a1.w};
    float hi[8] = {b0.x, b0.y, b0.z, b0.w, b1.x, b1.y, b1.z, b1.w};
    v16h r;
    #pragma unroll
    for (int i = 0; i < 8; i++) {
        r[i]     = (_Float16)(lo[i] * scale);
        r[8 + i] = (_Float16)(hi[i] * scale);
    }
    return r;
}

// ---------------- zero fill ----------------
__global__ void hg_zero(float* p, long long n) {
    long long i = (long long)blockIdx.x * blockDim.x + threadIdx.x;
    long long stride = (long long)gridDim.x * blockDim.x;
    for (; i < n; i += stride) p[i] = 0.0f;
}

// ---------------- f32 -> f16 convert ----------------
__global__ void hg_cvt_f16(const float* __restrict__ in, _Float16* __restrict__ out, long long n) {
    long long i = (long long)blockIdx.x * blockDim.x + threadIdx.x;
    long long stride = (long long)gridDim.x * blockDim.x;
    for (; i < n; i += stride) out[i] = (_Float16)in[i];
}

// ---------------- pack SAGE weights: Wt[n][k] = concat(W_self, W_neigh)[k][n], f16, 128x256 ----
__global__ void hg_pack_w(const float* __restrict__ Wself, const float* __restrict__ Wneigh,
                          _Float16* __restrict__ Wt) {
    int idx = blockIdx.x * blockDim.x + threadIdx.x;   // 32768 = 128*256
    int n = idx >> 8;
    int k = idx & 255;
    float v = (k < EM) ? Wself[k * EM + n] : Wneigh[(k - EM) * EM + n];
    Wt[n * 256 + k] = (_Float16)v;
}

// ---------------- edge scatter: msg_sum += x[src], deg += 1 (at dst) ----------------
__global__ void hg_edge_scatter(const float* __restrict__ x, const int* __restrict__ src,
                                const int* __restrict__ dst, float* msg, float* deg) {
    long long gid = (long long)blockIdx.x * blockDim.x + threadIdx.x;
    int e = (int)(gid >> 5);
    if (e >= NE) return;
    int lane = (int)(gid & 31);
    int s = src[e], d = dst[e];
    const float4 v = ((const float4*)(x + (long long)s * EM))[lane];
    float* md = msg + (long long)d * EM + lane * 4;
    atomicAdd(md + 0, v.x);
    atomicAdd(md + 1, v.y);
    atomicAdd(md + 2, v.z);
    atomicAdd(md + 3, v.w);
    if (lane == 0) atomicAdd(deg + d, 1.0f);
}

// ---------------- SAGE layer: h1 = relu([x, msg/deg] @ Wt^T + b) via WMMA ----------------
// one wave per 16x128 output row-block: A frag loaded once per K-step, 8 WMMAs reuse it.
__global__ void hg_sage_wmma(const _Float16* __restrict__ xh, const float* __restrict__ msg,
                             const float* __restrict__ deg, const _Float16* __restrict__ Wt,
                             const float* __restrict__ bsage, float* __restrict__ h1) {
    int wid = blockIdx.x * 8 + (threadIdx.x >> 5);
    if (wid >= 3125) return;                 // wave-uniform exit
    int lane = threadIdx.x & 31;
    int r = lane & 15, h = lane >> 4;
    int row = wid * 16 + r;
    float dinv = 1.0f / fmaxf(deg[row], 1.0f);

    v8f c[8] = {};
    #pragma unroll
    for (int s = 0; s < 8; s++) {
        int kb = (s & 3) * 32;               // K offset inside x / msg (each 128 wide)
        v16h a = (s < 4) ? load_frag_f16(xh + (long long)row * EM + kb, h)
                         : load_frag_f32(msg + (long long)row * EM + kb, h, dinv);
        #pragma unroll
        for (int nt = 0; nt < 8; nt++) {
            v16h b = load_frag_f16(Wt + (nt * 16 + r) * 256 + s * 32, h);
            c[nt] = __builtin_amdgcn_wmma_f32_16x16x32_f16(false, a, false, b,
                                                           (short)0, c[nt], false, false);
        }
    }

    #pragma unroll
    for (int nt = 0; nt < 8; nt++) {
        int col = nt * 16 + r;
        float bias = bsage[col];
        #pragma unroll
        for (int i = 0; i < 8; i++) {
            int m = wid * 16 + i + 8 * h;
            h1[(long long)m * EM + col] = fmaxf(c[nt][i] + bias, 0.0f);
        }
    }
}

// ---------------- gather + masked mean ----------------
__global__ void hg_gather(const float* __restrict__ x, const float* __restrict__ h1,
                          const int* __restrict__ seq, const int* __restrict__ user,
                          const int* __restrict__ mask, float* __restrict__ node_e,
                          float* __restrict__ user_e, float* __restrict__ hs) {
    int b = blockIdx.x, t = threadIdx.x;  // 128 threads
    float acc = 0.0f, cnt = 0.0f;
    for (int l = 0; l < SL; l++) {
        int node = seq[b * SL + l];
        float v = 0.5f * (h1[(long long)node * EM + t] + x[(long long)node * EM + t]);
        node_e[((long long)b * SL + l) * EM + t] = v;
        float mm = (float)mask[b * SL + l];
        acc += v * mm;
        cnt += mm;
    }
    hs[b * EM + t] = acc / cnt;
    int un = user[b] + ITEM_NUM;
    user_e[b * EM + t] = 0.5f * (h1[(long long)un * EM + t] + x[(long long)un * EM + t]);
}

// ---------------- dual attention: sess_vec, sess_user ----------------
__global__ void hg_attention(const float* __restrict__ node_e, const float* __restrict__ hs,
                             const float* __restrict__ user_e, const int* __restrict__ mask,
                             const int* __restrict__ pos_idx, const float* __restrict__ pos_w,
                             const float* __restrict__ w1, const float* __restrict__ w2,
                             const float* __restrict__ g1W, const float* __restrict__ g1b,
                             const float* __restrict__ g2W, const float* __restrict__ w3,
                             const float* __restrict__ w4, const float* __restrict__ g3W,
                             const float* __restrict__ g3b, const float* __restrict__ g4W,
                             float* __restrict__ sess_vec, float* __restrict__ sess_user) {
    int b = blockIdx.x, t = threadIdx.x;  // 128 threads
    __shared__ float vec[2 * EM];
    __shared__ float hsv[EM], usv[EM], hsg[EM], ug[EM];
    __shared__ float t0[EM], t1[EM];

    hsv[t] = hs[b * EM + t];
    usv[t] = user_e[b * EM + t];
    __syncthreads();
    {
        float a1 = 0.0f, a2 = 0.0f;
        for (int k = 0; k < EM; k++) {
            a1 = fmaf(hsv[k], g2W[k * EM + t], a1);
            a2 = fmaf(usv[k], g4W[k * EM + t], a2);
        }
        hsg[t] = a1 + g1b[t];
        ug[t]  = a2 + g3b[t];
    }
    float sv = 0.0f, su = 0.0f;

    for (int l = 0; l < SL; l++) {
        __syncthreads();
        int pi = pos_idx[b * SL + l];
        vec[t] = pos_w[pi * EM + t];
        vec[EM + t] = node_e[((long long)b * SL + l) * EM + t];
        float mval = (float)mask[b * SL + l];
        __syncthreads();

        // nh = tanh(concat @ w_1)
        float acc = 0.0f;
        for (int k = 0; k < 2 * EM; k++) acc = fmaf(vec[k], w1[k * EM + t], acc);
        t0[t] = tanhf(acc);
        __syncthreads();

        // s1 = sigmoid(nh @ glu1_W + glu1_b + hs @ glu2_W)
        acc = 0.0f;
        for (int k = 0; k < EM; k++) acc = fmaf(t0[k], g1W[k * EM + t], acc);
        t1[t] = sigmoidf_(acc + hsg[t]);
        __syncthreads();

        float beta = 0.0f;
        for (int k = 0; k < EM; k++) beta = fmaf(t1[k], w2[k], beta);
        beta *= mval;
        sv = fmaf(beta, vec[EM + t], sv);

        // nh2 = tanh(node @ w_3)
        acc = 0.0f;
        for (int k = 0; k < EM; k++) acc = fmaf(vec[EM + k], w3[k * EM + t], acc);
        __syncthreads();
        t0[t] = tanhf(acc);
        __syncthreads();

        acc = 0.0f;
        for (int k = 0; k < EM; k++) acc = fmaf(t0[k], g3W[k * EM + t], acc);
        t1[t] = sigmoidf_(acc + ug[t]);
        __syncthreads();

        float beta2 = 0.0f;
        for (int k = 0; k < EM; k++) beta2 = fmaf(t1[k], w4[k], beta2);
        beta2 *= mval;
        su = fmaf(beta2, vec[EM + t], su);
    }
    sess_vec[b * EM + t] = sv;
    sess_user[b * EM + t] = su;
}

// ---------------- alpha gate + seq_embeds (stored f16 for WMMA) ----------------
__global__ void hg_combine(const float* __restrict__ sv, const float* __restrict__ su,
                           const float* __restrict__ ue, const float* __restrict__ scW,
                           const float* __restrict__ scb, _Float16* __restrict__ seq_h) {
    int b = blockIdx.x, t = threadIdx.x;  // 128 threads
    float a = scb[0];
    for (int k = 0; k < EM; k++) a = fmaf(sv[b * EM + k], scW[k], a);
    for (int k = 0; k < EM; k++) a = fmaf(su[b * EM + k], scW[EM + k], a);
    float alpha = sigmoidf_(a);
    float v = ue[b * EM + t] + alpha * sv[b * EM + t] + (1.0f - alpha) * su[b * EM + t];
    seq_h[b * EM + t] = (_Float16)v;
}

// ---------------- scores = seq_embeds @ v2e.T via WMMA ----------------
// one wave per 4 M-tiles x 1 N-tile: the big-matrix (Xh) fragment is loaded once, reused 4x.
// 16 M-groups * 3125 N-tiles = 50000 waves = 6250 blocks of 8 waves, exact.
__global__ void hg_scores_wmma(const _Float16* __restrict__ S, const _Float16* __restrict__ Xh,
                               float* __restrict__ out) {
    int wid = blockIdx.x * 8 + (threadIdx.x >> 5);
    int lane = threadIdx.x & 31;
    int tg = wid / 3125;     // M group (0..15), 4 tiles of 16 rows each
    int tn = wid % 3125;     // N tile
    int r = lane & 15, h = lane >> 4;
    int col = tn * 16 + r;
    int row0 = tg * 64 + r;

    v8f c0 = {}, c1 = {}, c2 = {}, c3 = {};
    #pragma unroll
    for (int s = 0; s < 4; s++) {
        int kb = s * 32;
        v16h b  = load_frag_f16(Xh + (long long)col * EM + kb, h);
        v16h a0 = load_frag_f16(S + (row0 + 0)  * EM + kb, h);
        c0 = __builtin_amdgcn_wmma_f32_16x16x32_f16(false, a0, false, b, (short)0, c0, false, false);
        v16h a1 = load_frag_f16(S + (row0 + 16) * EM + kb, h);
        c1 = __builtin_amdgcn_wmma_f32_16x16x32_f16(false, a1, false, b, (short)0, c1, false, false);
        v16h a2 = load_frag_f16(S + (row0 + 32) * EM + kb, h);
        c2 = __builtin_amdgcn_wmma_f32_16x16x32_f16(false, a2, false, b, (short)0, c2, false, false);
        v16h a3 = load_frag_f16(S + (row0 + 48) * EM + kb, h);
        c3 = __builtin_amdgcn_wmma_f32_16x16x32_f16(false, a3, false, b, (short)0, c3, false, false);
    }

    int n = tn * 16 + (lane & 15);
    #pragma unroll
    for (int i = 0; i < 8; i++) {
        int m = tg * 64 + i + 8 * h;
        out[(long long)(m)      * NNODES + n] = c0[i];
        out[(long long)(m + 16) * NNODES + n] = c1[i];
        out[(long long)(m + 32) * NNODES + n] = c2[i];
        out[(long long)(m + 48) * NNODES + n] = c3[i];
    }
}

extern "C" void kernel_launch(void* const* d_in, const int* in_sizes, int n_in,
                              void* d_out, int out_size, void* d_ws, size_t ws_size,
                              hipStream_t stream) {
    const float* v2e    = (const float*)d_in[0];
    const float* pos_w  = (const float*)d_in[1];
    const float* W_self = (const float*)d_in[2];
    const float* W_neigh= (const float*)d_in[3];
    const float* b_sage = (const float*)d_in[4];
    const float* w_1    = (const float*)d_in[5];
    const float* w_2    = (const float*)d_in[6];
    const float* glu1_W = (const float*)d_in[7];
    const float* glu1_b = (const float*)d_in[8];
    const float* glu2_W = (const float*)d_in[9];
    const float* w_3    = (const float*)d_in[10];
    const float* w_4    = (const float*)d_in[11];
    const float* glu3_W = (const float*)d_in[12];
    const float* glu3_b = (const float*)d_in[13];
    const float* glu4_W = (const float*)d_in[14];
    const float* sc_W   = (const float*)d_in[15];
    const float* sc_b   = (const float*)d_in[16];
    const int* src      = (const int*)d_in[17];
    const int* dst      = (const int*)d_in[18];
    const int* user     = (const int*)d_in[19];
    const int* seq      = (const int*)d_in[20];
    const int* mask     = (const int*)d_in[21];
    const int* pos_idx  = (const int*)d_in[22];
    float* out = (float*)d_out;

    // workspace carve-up (256-byte aligned)
    char* ws = (char*)d_ws;
    size_t off = 0;
    auto take = [&](size_t bytes) {
        char* p = ws + off;
        off = (off + bytes + 255) & ~(size_t)255;
        return p;
    };
    float*     msg    = (float*)take((size_t)NNODES * EM * 4);
    float*     deg    = (float*)take((size_t)NNODES * 4);
    float*     h1     = (float*)take((size_t)NNODES * EM * 4);
    _Float16*  xh     = (_Float16*)take((size_t)NNODES * EM * 2);
    _Float16*  Wt     = (_Float16*)take((size_t)EM * 2 * EM * 2);
    float*     node_e = (float*)take((size_t)BS * SL * EM * 4);
    float*     user_e = (float*)take((size_t)BS * EM * 4);
    float*     hs     = (float*)take((size_t)BS * EM * 4);
    float*     sess_v = (float*)take((size_t)BS * EM * 4);
    float*     sess_u = (float*)take((size_t)BS * EM * 4);
    _Float16*  seq_h  = (_Float16*)take((size_t)BS * EM * 2);
    (void)ws_size; (void)in_sizes; (void)n_in; (void)out_size;

    // 1) zero accumulators
    hg_zero<<<2048, 256, 0, stream>>>(msg, (long long)NNODES * EM);
    hg_zero<<<256, 256, 0, stream>>>(deg, (long long)NNODES);

    // 2) edge scatter (32 lanes per edge)
    hg_edge_scatter<<<(NE * 32) / 256, 256, 0, stream>>>(v2e, src, dst, msg, deg);

    // 3) f16 copy of embedding table (A for SAGE, B for scores) + packed transposed weights
    hg_cvt_f16<<<2048, 256, 0, stream>>>(v2e, xh, (long long)NNODES * EM);
    hg_pack_w<<<(EM * 2 * EM) / 256, 256, 0, stream>>>(W_self, W_neigh, Wt);

    // 4) SAGE layer GEMM (WMMA): 3125 row-block waves, 8 per block
    hg_sage_wmma<<<(3125 + 7) / 8, 256, 0, stream>>>(xh, msg, deg, Wt, b_sage, h1);

    // 5) gather sequence/user embeddings + masked mean
    hg_gather<<<BS, EM, 0, stream>>>(v2e, h1, seq, user, mask, node_e, user_e, hs);

    // 6) dual-branch attention
    hg_attention<<<BS, EM, 0, stream>>>(node_e, hs, user_e, mask, pos_idx, pos_w,
                                        w_1, w_2, glu1_W, glu1_b, glu2_W,
                                        w_3, w_4, glu3_W, glu3_b, glu4_W,
                                        sess_v, sess_u);

    // 7) gate + final session embedding (f16)
    hg_combine<<<BS, EM, 0, stream>>>(sess_v, sess_u, user_e, sc_W, sc_b, seq_h);

    // 8) scores GEMM (WMMA): 50000 waves, 8 per block
    hg_scores_wmma<<<6250, 256, 0, stream>>>(seq_h, xh, out);
}
